// HierarchicalAttention_21663815041759
// MI455X (gfx1250) — matrix operand order repaired and checked by reading, not compiled, not run on hardware
//
#include <hip/hip_runtime.h>
#include <cstdint>
#include <cstddef>

#define DIM   768
#define HEADS 12
#define HD    64
#define WS    16
#define CT    4
#define BATCH 2
#define SEQ   2048
#define NW    (SEQ / WS)        // 128 windows per batch
#define NTOT  (SEQ + NW * CT)   // 2560 combined tokens
#define BN    (BATCH * NTOT)    // 5120 rows into QKV GEMM
#define QKVN  (3 * DIM)         // 2304
#define GK    768               // shared inner dimension of all three GEMMs
#define ATT_SCALE 0.125f        // 64^-0.5
#define LNEPS 1e-5f

typedef __attribute__((ext_vector_type(16))) _Float16 v16h;
typedef __attribute__((ext_vector_type(8)))  _Float16 v8h;
typedef __attribute__((ext_vector_type(8)))  float    v8f;

// ---- WMMA fragment loaders (wave32, 16x16x32 f16) -------------------------
// A (16xK tile, row-major storage [rows][ld]): lane&15 = row,
//   K elements: 16*(lane>>4) + e  (e = 0..15 contiguous)
__device__ __forceinline__ v16h frag_a_ptr(const _Float16* rowptr, int k0) {
  int lane = threadIdx.x & 31;
  const _Float16* p = rowptr + k0 + ((lane >> 4) << 4);
  v8h lo = *reinterpret_cast<const v8h*>(p);
  v8h hi = *reinterpret_cast<const v8h*>(p + 8);
  v16h r;
#pragma unroll
  for (int i = 0; i < 8; ++i) { r[i] = lo[i]; r[i + 8] = hi[i]; }
  return r;
}
__device__ __forceinline__ v16h frag_a(const _Float16* base, int ld, int row0, int k0) {
  int lane = threadIdx.x & 31;
  return frag_a_ptr(base + (size_t)(row0 + (lane & 15)) * ld, k0);
}
// B (Kx16 tile, stored column-major as rows: colptr -> 16 contiguous K values)
//   lane&15 = col, K elements: 16*(lane>>4) + e  (e = 0..15 contiguous)
__device__ __forceinline__ v16h frag_b_ptr(const _Float16* colptr, int k0) {
  int lane = threadIdx.x & 31;
  const _Float16* p = colptr + k0 + ((lane >> 4) << 4);
  v8h lo = *reinterpret_cast<const v8h*>(p);
  v8h hi = *reinterpret_cast<const v8h*>(p + 8);
  v16h r;
#pragma unroll
  for (int i = 0; i < 8; ++i) { r[i] = lo[i]; r[i + 8] = hi[i]; }
  return r;
}
__device__ __forceinline__ v16h frag_b(const _Float16* baseT, int ld, int col0, int k0) {
  int lane = threadIdx.x & 31;
  return frag_b_ptr(baseT + (size_t)(col0 + (lane & 15)) * ld, k0);
}

// ---- weight transpose + fp16 cast: W[K][N] -> Wt[N][K] --------------------
__global__ void k_wt(const float* __restrict__ W, _Float16* __restrict__ Wt, int K, int N) {
  int i = blockIdx.x * blockDim.x + threadIdx.x;
  if (i >= K * N) return;
  int k = i / N, n = i % N;
  Wt[(size_t)n * K + k] = (_Float16)W[i];
}

// ---- per-window mean ------------------------------------------------------
__global__ __launch_bounds__(256) void k_mean(const float* __restrict__ x,
                                              _Float16* __restrict__ meanh) {
  int g = blockIdx.x;                      // b*NW + w
  int b = g / NW, w = g % NW;
  const float* base = x + ((size_t)b * SEQ + w * WS) * DIM;
#pragma unroll
  for (int i = 0; i < 3; ++i) {
    int c = threadIdx.x + i * 256;
    float s = 0.f;
#pragma unroll
    for (int t = 0; t < WS; ++t) s += base[(size_t)t * DIM + c];
    meanh[(size_t)g * DIM + c] = (_Float16)(s * (1.f / WS));
  }
}

// ---- WMMA GEMM, K fixed at 768: C[M][N] = A[M][768] * Bt[N][768]^T + bias -
// block = 128 threads (4 waves); all waves share m0, so the contiguous
// 16x768 fp16 A strip is staged ONCE per block into LDS with the CDNA5
// async-to-LDS engine (ASYNCcnt), then read back as ds_read fragments.
// The k-loop is unrolled by 2 with ping-pong B fragment buffers so the
// next chunk's global loads issue before the current chunk's WMMAs, with
// no register-rotation moves.
__global__ __launch_bounds__(128) void k_gemm(const _Float16* __restrict__ A,
                                              const _Float16* __restrict__ Bt,
                                              const float* __restrict__ bias,
                                              float* __restrict__ C,
                                              int M, int N) {
  __shared__ __align__(16) _Float16 As[16 * GK];   // 24 KB
  int m0  = blockIdx.x * 16;
  int tid = threadIdx.x;
  int n0  = blockIdx.y * 256 + (tid >> 5) * 64;

  // ---- async stage of the A strip (contiguous 16*768 halves = 1536 x 16B) --
  {
    const _Float16* Ab = A + (size_t)m0 * GK;
    unsigned ldsbase = (unsigned)(uintptr_t)(&As[0]);  // LDS offset = addr[31:0]
#pragma unroll
    for (int i = 0; i < 12; ++i) {
      int chunk = tid + i * 128;                       // 16-byte chunk index
      unsigned loff = ldsbase + chunk * 16;
      const _Float16* g = Ab + chunk * 8;
      asm volatile("global_load_async_to_lds_b128 %0, %1, off"
                   :: "v"(loff), "v"(g) : "memory");
    }
    asm volatile("s_wait_asynccnt 0x0" ::: "memory");
  }
  __syncthreads();

  v8f  acc[4] = {};
  v16h b0[4], b1[4];
#pragma unroll
  for (int t = 0; t < 4; ++t) b0[t] = frag_b(Bt, GK, n0 + t * 16, 0);

  for (int k0 = 0; k0 < GK; k0 += 64) {             // 12 iterations, ping-pong
    // prefetch odd chunk while even chunk computes
#pragma unroll
    for (int t = 0; t < 4; ++t) b1[t] = frag_b(Bt, GK, n0 + t * 16, k0 + 32);
    v16h a0 = frag_a(As, GK, 0, k0);                // ds_read from LDS
#pragma unroll
    for (int t = 0; t < 4; ++t)
      acc[t] = __builtin_amdgcn_wmma_f32_16x16x32_f16(false, a0, false, b0[t],
                                                      (short)0, acc[t], false, false);
    // prefetch next even chunk while odd chunk computes
    if (k0 + 64 < GK) {
#pragma unroll
      for (int t = 0; t < 4; ++t) b0[t] = frag_b(Bt, GK, n0 + t * 16, k0 + 64);
    }
    v16h a1 = frag_a(As, GK, 0, k0 + 32);
#pragma unroll
    for (int t = 0; t < 4; ++t)
      acc[t] = __builtin_amdgcn_wmma_f32_16x16x32_f16(false, a1, false, b1[t],
                                                      (short)0, acc[t], false, false);
  }

  int col = (tid & 31) & 15;
  int mb  = ((tid & 31) >> 4) * 8;
#pragma unroll
  for (int t = 0; t < 4; ++t) {
    int n = n0 + t * 16 + col;
    float bv = bias[n];
#pragma unroll
    for (int r = 0; r < 8; ++r)
      C[(size_t)(m0 + mb + r) * N + n] = acc[t][r] + bv;
  }
}

// ---- fused LayerNorm over combined sequence (x rows + carrier rows) -------
__global__ __launch_bounds__(256) void k_ln(const float* __restrict__ x,
                                            const float* __restrict__ car,
                                            const float* __restrict__ gamma,
                                            const float* __restrict__ beta,
                                            _Float16* __restrict__ lnh) {
  __shared__ float red[256];
  int g = blockIdx.x;                       // b*NTOT + t
  int b = g / NTOT, t = g % NTOT;
  const float* src = (t < SEQ) ? (x + ((size_t)b * SEQ + t) * DIM)
                               : (car + ((size_t)b * NW + (t - SEQ) / CT) * DIM);
  int tid = threadIdx.x;
  float v[3]; float lsum = 0.f;
#pragma unroll
  for (int i = 0; i < 3; ++i) { v[i] = src[tid + i * 256]; lsum += v[i]; }
  red[tid] = lsum; __syncthreads();
  for (int s = 128; s > 0; s >>= 1) { if (tid < s) red[tid] += red[tid + s]; __syncthreads(); }
  float mu = red[0] * (1.f / DIM);
  __syncthreads();
  float lvar = 0.f;
#pragma unroll
  for (int i = 0; i < 3; ++i) { float d = v[i] - mu; lvar += d * d; }
  red[tid] = lvar; __syncthreads();
  for (int s = 128; s > 0; s >>= 1) { if (tid < s) red[tid] += red[tid + s]; __syncthreads(); }
  float rstd = rsqrtf(red[0] * (1.f / DIM) + LNEPS);
#pragma unroll
  for (int i = 0; i < 3; ++i) {
    int c = tid + i * 256;
    lnh[(size_t)g * DIM + c] = (_Float16)((v[i] - mu) * rstd * gamma[c] + beta[c]);
  }
}

// ---- split QKV rows into head layouts: Q (pre-scaled), K row-major, V^T ---
__global__ __launch_bounds__(256) void k_split(const float* __restrict__ qkv,
                                               _Float16* __restrict__ Qh,
                                               _Float16* __restrict__ Kh,
                                               _Float16* __restrict__ Vth) {
  int g = blockIdx.x;                       // b*NTOT + t
  int b = g / NTOT, t = g % NTOT;
  const float* row = qkv + (size_t)g * QKVN;
#pragma unroll
  for (int i = 0; i < 3; ++i) {
    int c = threadIdx.x + i * 256;
    int h = c >> 6, d = c & 63;
    size_t qk = ((size_t)(b * HEADS + h) * NTOT + t) * HD + d;
    Qh[qk] = (_Float16)(row[c] * ATT_SCALE);
    Kh[qk] = (_Float16)(row[DIM + c]);
    Vth[((size_t)(b * HEADS + h) * HD + d) * NTOT + t] = (_Float16)(row[2 * DIM + c]);
  }
}

// ---- window attention: one wave per (b, head, window) ---------------------
// 16 queries x 20 keys (16 window tokens + 4 carriers of the same window).
__global__ __launch_bounds__(32) void k_winattn(const _Float16* __restrict__ Qh,
                                                const _Float16* __restrict__ Kh,
                                                const _Float16* __restrict__ Vth,
                                                _Float16* __restrict__ Oh) {
  __shared__ float sm[16][32];
  __shared__ _Float16 pm[16][32];
  int id = blockIdx.x;
  int w = id % NW;
  int h = (id / NW) % HEADS;
  int b = id / (NW * HEADS);
  int lane = threadIdx.x;

  const _Float16* Qbase = Qh + ((size_t)(b * HEADS + h) * NTOT + w * WS) * HD;
  const _Float16* Kbase = Kh + (size_t)(b * HEADS + h) * NTOT * HD;

  // S = Q K^T : tile0 = window keys 0..15, tile1 = carrier keys (cols 16..19)
  v8f s0 = {}; v8f s1 = {};
#pragma unroll
  for (int k0 = 0; k0 < HD; k0 += 32) {
    v16h a  = frag_a(Qbase, HD, 0, k0);
    v16h b0 = frag_b(Kbase, HD, w * WS, k0);
    int cn   = lane & 15;
    int ctok = SEQ + w * CT + (cn & 3);          // cols 4..15 are dupes, masked below
    v16h b1 = frag_b_ptr(Kbase + (size_t)ctok * HD, k0);
    s0 = __builtin_amdgcn_wmma_f32_16x16x32_f16(false, a, false, b0, (short)0, s0, false, false);
    s1 = __builtin_amdgcn_wmma_f32_16x16x32_f16(false, a, false, b1, (short)0, s1, false, false);
  }
  {
    int col = lane & 15, mb = (lane >> 4) * 8;
#pragma unroll
    for (int r = 0; r < 8; ++r) {
      sm[mb + r][col]      = s0[r];
      sm[mb + r][16 + col] = s1[r];
    }
  }
  __syncthreads();
  if (lane < 16) {                      // softmax over the 20 valid keys
    float mx = -1e30f;
    for (int c = 0; c < 20; ++c) mx = fmaxf(mx, sm[lane][c]);
    float sum = 0.f;
    for (int c = 0; c < 20; ++c) { float e = __expf(sm[lane][c] - mx); sm[lane][c] = e; sum += e; }
    float inv = 1.f / sum;
    for (int c = 0; c < 32; ++c)
      pm[lane][c] = (c < 20) ? (_Float16)(sm[lane][c] * inv) : (_Float16)0.f;
  }
  __syncthreads();

  // O = P V : K-dim 32 covers all 20 keys (zero-padded probabilities)
  v16h pa = frag_a(&pm[0][0], 32, 0, 0);
  const _Float16* Vbase = Vth + (size_t)(b * HEADS + h) * HD * NTOT;
  int col = lane & 15, mb = (lane >> 4) * 8;
#pragma unroll
  for (int t = 0; t < 4; ++t) {
    int d = t * 16 + col;
    const _Float16* vrow = Vbase + (size_t)d * NTOT;
    int kb = (lane >> 4) * 16;
    v16h bf;
#pragma unroll
    for (int i = 0; i < 16; ++i) {
      int k = kb + i;
      int tok = (k < 16) ? (w * WS + k)
              : (k < 20) ? (SEQ + w * CT + (k - 16))
                         : (SEQ + w * CT + 3);   // prob == 0 there
      bf[i] = vrow[tok];
    }
    v8f o = {};
    o = __builtin_amdgcn_wmma_f32_16x16x32_f16(false, pa, false, bf, (short)0, o, false, false);
#pragma unroll
    for (int r = 0; r < 8; ++r) {
      int m = mb + r;
      Oh[((size_t)b * SEQ + w * WS + m) * DIM + h * HD + t * 16 + col] = (_Float16)o[r];
    }
  }
}

extern "C" void kernel_launch(void* const* d_in, const int* in_sizes, int n_in,
                              void* d_out, int out_size, void* d_ws, size_t ws_size,
                              hipStream_t stream) {
  (void)in_sizes; (void)n_in; (void)out_size; (void)ws_size;
  const float* x      = (const float*)d_in[0];
  const float* W_qkv  = (const float*)d_in[1];
  const float* b_qkv  = (const float*)d_in[2];
  const float* W_car  = (const float*)d_in[3];
  const float* b_car  = (const float*)d_in[4];
  const float* W_proj = (const float*)d_in[5];
  const float* b_proj = (const float*)d_in[6];
  const float* gamma  = (const float*)d_in[7];
  const float* beta   = (const float*)d_in[8];
  float* out = (float*)d_out;

  char* wsp = (char*)d_ws;
  size_t off = 0;
  auto walloc = [&](size_t bytes) -> void* {
    void* p = wsp + off;
    off += (bytes + 255) & ~(size_t)255;
    return p;
  };
  _Float16* WqkvT  = (_Float16*)walloc((size_t)DIM * QKVN * 2);
  _Float16* WcarT  = (_Float16*)walloc((size_t)DIM * DIM * 2);
  _Float16* WprojT = (_Float16*)walloc((size_t)DIM * DIM * 2);
  _Float16* meanh  = (_Float16*)walloc((size_t)BATCH * NW * DIM * 2);
  float*    car    = (float*)   walloc((size_t)BATCH * NW * DIM * 4);
  _Float16* lnh    = (_Float16*)walloc((size_t)BN * DIM * 2);
  float*    qkv    = (float*)   walloc((size_t)BN * QKVN * 4);
  _Float16* Qh     = (_Float16*)walloc((size_t)BATCH * HEADS * NTOT * HD * 2);
  _Float16* Kh     = (_Float16*)walloc((size_t)BATCH * HEADS * NTOT * HD * 2);
  _Float16* Vth    = (_Float16*)walloc((size_t)BATCH * HEADS * HD * NTOT * 2);
  _Float16* attnh  = (_Float16*)walloc((size_t)BATCH * SEQ * DIM * 2);

  // 1) weights -> fp16 transposed [N][K]
  { int n = DIM * QKVN; k_wt<<<(n + 255) / 256, 256, 0, stream>>>(W_qkv,  WqkvT,  DIM, QKVN); }
  { int n = DIM * DIM;  k_wt<<<(n + 255) / 256, 256, 0, stream>>>(W_car,  WcarT,  DIM, DIM);  }
  { int n = DIM * DIM;  k_wt<<<(n + 255) / 256, 256, 0, stream>>>(W_proj, WprojT, DIM, DIM);  }
  // 2) per-window means
  k_mean<<<BATCH * NW, 256, 0, stream>>>(x, meanh);
  // 3) carrier tokens: mean @ W_car + b_car   (M=256, N=768, K=768)
  k_gemm<<<dim3((BATCH * NW) / 16, DIM / 256), 128, 0, stream>>>(meanh, WcarT, b_car, car,
                                                                 BATCH * NW, DIM);
  // 4) LayerNorm(combined) -> fp16            (5120 rows)
  k_ln<<<BN, 256, 0, stream>>>(x, car, gamma, beta, lnh);
  // 5) QKV GEMM                               (M=5120, N=2304, K=768)
  k_gemm<<<dim3(BN / 16, QKVN / 256), 128, 0, stream>>>(lnh, WqkvT, b_qkv, qkv, BN, QKVN);
  // 6) split into head layouts (Q pre-scaled)
  k_split<<<BN, 256, 0, stream>>>(qkv, Qh, Kh, Vth);
  // 7) window-token attention (carrier-query outputs are sliced away by the
  //    reference, so only window queries are computed)
  k_winattn<<<BATCH * HEADS * NW, 32, 0, stream>>>(Qh, Kh, Vth, attnh);
  // 8) output projection straight into d_out  (M=4096, N=768, K=768)
  k_gemm<<<dim3((BATCH * SEQ) / 16, DIM / 256), 128, 0, stream>>>(attnh, WprojT, b_proj, out,
                                                                  BATCH * SEQ, DIM);
}
